// Seq2Seq_14894946582554
// MI455X (gfx1250) — matrix operand order repaired and checked by reading, not compiled, not run on hardware
//
#include <hip/hip_runtime.h>
#include <stdint.h>

#define NB 32          // batch
#define LSRC 64        // source length
#define LTGT 48        // target length
#define ED 512         // embedding
#define HD 1024        // hidden
#define G4H 4096       // 4*H gate width
#define VTGT 32000     // target vocab

typedef __bf16 v16bf __attribute__((ext_vector_type(16)));
typedef float  v8f   __attribute__((ext_vector_type(8)));
typedef unsigned short u16;

// ---------- scalar helpers ----------
__device__ __forceinline__ u16 f2bf(float f) {
    union { float f; unsigned u; } x; x.f = f;
    unsigned r = x.u + 0x7FFFu + ((x.u >> 16) & 1u);   // round-to-nearest-even
    return (u16)(r >> 16);
}
__device__ __forceinline__ float bf2f(u16 u) {
    union { unsigned u; float f; } x; x.u = ((unsigned)u) << 16;
    return x.f;
}
__device__ __forceinline__ float sigm(float x) { return 1.0f / (1.0f + __expf(-x)); }

// ---------- WMMA fragment loads (wave32, 16x16x32 bf16) ----------
union FragU { uint4 q[2]; v16bf v; };

// A: row-major [M][ldk] bf16. Lane l (m = m0 + (l&15), half = l>>4) holds
// K runs [kb*32 + half*8, +8) and [kb*32 + 16 + half*8, +8) -> two b128 loads.
__device__ __forceinline__ v16bf load_frag_a(const u16* __restrict__ A, int ldk, int m0, int kb) {
    int lane = threadIdx.x & 31;
    int half = lane >> 4;
    int m = m0 + (lane & 15);
    const u16* p = A + (size_t)m * ldk + kb * 32 + half * 8;
    FragU u;
    u.q[0] = *(const uint4*)(p);        // K = base .. base+7
    u.q[1] = *(const uint4*)(p + 16);   // K = base+16 .. base+23
    return u.v;
}

// B: packed layout [ntile][KB][lane][16 bf16]; per lane 32 contiguous bytes.
// Element i = K(kb*32 + (i>>3)*16 + (lane>>4)*8 + (i&7)), col = ntile*16 + (lane&15).
__device__ __forceinline__ v16bf load_frag_b(const u16* __restrict__ Bp, int KB, int ntile, int kb) {
    int lane = threadIdx.x & 31;
    const u16* p = Bp + ((((size_t)ntile * KB + kb) * 32 + lane) << 4);
    FragU u;
    u.q[0] = *(const uint4*)(p);
    u.q[1] = *(const uint4*)(p + 8);
    return u.v;
}

#define WMMA_BF16(a, b, c) \
    __builtin_amdgcn_wmma_f32_16x16x32_bf16(false, (a), false, (b), (short)0, (c), false, false)

// ---------- weight repack: f32 [K][NC] row-major -> WMMA-packed bf16 ----------
__global__ __launch_bounds__(256) void pack_w(const float* __restrict__ W,
                                              u16* __restrict__ out, int K, int NC) {
    size_t idx = (size_t)blockIdx.x * 256 + threadIdx.x;
    size_t total = (size_t)K * NC;
    if (idx >= total) return;
    int i    = (int)(idx & 15);
    int lane = (int)((idx >> 4) & 31);
    size_t rest = idx >> 9;
    int KB = K >> 5;
    int kb = (int)(rest % KB);
    int nt = (int)(rest / KB);
    int col = nt * 16 + (lane & 15);
    int k   = kb * 32 + (i >> 3) * 16 + (lane >> 4) * 8 + (i & 7);
    out[idx] = f2bf(W[(size_t)k * NC + col]);
}

// ---------- source embedding gather -> bf16 [LSRC][NB][ED] ----------
__global__ __launch_bounds__(256) void embed_src_k(const int* __restrict__ src,
                                                   const float* __restrict__ emb,
                                                   u16* __restrict__ x_src) {
    size_t idx = (size_t)blockIdx.x * 256 + threadIdx.x;
    if (idx >= (size_t)LSRC * NB * ED) return;
    int e = (int)(idx % ED);
    int n = (int)((idx / ED) % NB);
    int t = (int)(idx / ((size_t)ED * NB));
    int tok = src[n * LSRC + t];
    x_src[idx] = f2bf(emb[(size_t)tok * ED + e]);
}

// ---------- fused LSTM step: gates = x@Wi + h@Wh + b -> pointwise -> h,c ----------
// grid = HD/32 = 32 blocks (32 hidden cols each), 256 threads = 8 waves:
// wave w -> gate = w>>1, column subtile = w&1; each wave computes BOTH 16-row
// M-tiles with two accumulators, reusing every B (weight) fragment twice.
__global__ __launch_bounds__(256) void lstm_step(
        const u16* __restrict__ x,     // [NB][Kx] bf16
        const u16* __restrict__ h_in,  // [NB][HD] bf16
        const u16* __restrict__ Wi_p,  // packed, NC=4096, K=Kx
        const u16* __restrict__ Wh_p,  // packed, NC=4096, K=HD
        const float* __restrict__ b,   // [4H]
        float* __restrict__ c,         // [NB][HD] f32 (in/out)
        u16* __restrict__ h_out,       // [NB][HD] bf16
        u16* __restrict__ h_copy,      // optional second copy (enc_h[t]) or null
        int Kx) {
    int tid  = threadIdx.x;
    int lane = tid & 31;
    int w    = tid >> 5;
    int gate = w >> 1;                       // 0..3 : i,f,g,o
    int cs   = w & 1;                        // column subtile within block
    int blk  = blockIdx.x;                   // 32-column tile, 0..31
    int ct   = blk * 2 + cs;                 // 16-col tile index within HD
    int gntile = gate * (HD / 16) + ct;      // ntile within 4H

    float bias = b[gate * HD + ct * 16 + (lane & 15)];
    v8f acc0, acc1;
    #pragma unroll
    for (int i = 0; i < 8; ++i) { acc0[i] = bias; acc1[i] = bias; }

    int KBx = Kx >> 5;
    for (int kb = 0; kb < KBx; ++kb) {
        v16bf bb = load_frag_b(Wi_p, KBx, gntile, kb);
        v16bf a0 = load_frag_a(x, Kx, 0,  kb);
        v16bf a1 = load_frag_a(x, Kx, 16, kb);
        acc0 = WMMA_BF16(a0, bb, acc0);
        acc1 = WMMA_BF16(a1, bb, acc1);
    }
    for (int kb = 0; kb < (HD >> 5); ++kb) {
        v16bf bb = load_frag_b(Wh_p, HD >> 5, gntile, kb);
        v16bf a0 = load_frag_a(h_in, HD, 0,  kb);
        v16bf a1 = load_frag_a(h_in, HD, 16, kb);
        acc0 = WMMA_BF16(a0, bb, acc0);
        acc1 = WMMA_BF16(a1, bb, acc1);
    }

    __shared__ float g4[4][NB][32];   // 16 KB
    int mrow = (lane >> 4) * 8;       // C layout: lanes16-31 hold rows 8..15
    int jc   = cs * 16 + (lane & 15);
    #pragma unroll
    for (int i = 0; i < 8; ++i) {
        g4[gate][mrow + i][jc]      = acc0[i];
        g4[gate][16 + mrow + i][jc] = acc1[i];
    }
    __syncthreads();

    for (int e = tid; e < NB * 32; e += 256) {
        int m = e >> 5, j = e & 31;
        int col = blk * 32 + j;
        float ig = sigm(g4[0][m][j]);
        float fg = sigm(g4[1][m][j]);
        float gg = tanhf(g4[2][m][j]);
        float og = sigm(g4[3][m][j]);
        float cn = fg * c[(size_t)m * HD + col] + ig * gg;
        float hn = og * tanhf(cn);
        c[(size_t)m * HD + col] = cn;
        u16 hb = f2bf(hn);
        h_out[(size_t)m * HD + col] = hb;
        if (h_copy) h_copy[(size_t)m * HD + col] = hb;
    }
}

// ---------- generic WMMA GEMM: C[M][NC] = A[M][K](bf16) @ Bp + bias ----------
// M must be a multiple of 32: each wave computes a PAIR of M-tiles for one
// N-tile, sharing every B fragment across two WMMAs.
__global__ __launch_bounds__(256) void gemm_bf16(
        const u16* __restrict__ A, int M, int K,
        const u16* __restrict__ Bp, int NC,
        const float* __restrict__ bias, float* __restrict__ Cout) {
    int w = threadIdx.x >> 5;
    int lane = threadIdx.x & 31;
    size_t id = (size_t)blockIdx.x * 8 + w;
    int NT = NC >> 4;
    int MP = M >> 5;                  // pairs of 16-row tiles
    if (id >= (size_t)MP * NT) return;
    int mp = (int)(id / NT);
    int nt = (int)(id % NT);

    float bv = bias ? bias[nt * 16 + (lane & 15)] : 0.0f;
    v8f acc0, acc1;
    #pragma unroll
    for (int i = 0; i < 8; ++i) { acc0[i] = bv; acc1[i] = bv; }

    int KB = K >> 5;
    for (int kb = 0; kb < KB; ++kb) {
        v16bf bb = load_frag_b(Bp, KB, nt, kb);
        v16bf a0 = load_frag_a(A, K, mp * 32,      kb);
        v16bf a1 = load_frag_a(A, K, mp * 32 + 16, kb);
        acc0 = WMMA_BF16(a0, bb, acc0);
        acc1 = WMMA_BF16(a1, bb, acc1);
    }
    int r0  = mp * 32 + ((lane >> 4) * 8);
    int col = nt * 16 + (lane & 15);
    #pragma unroll
    for (int i = 0; i < 8; ++i) {
        Cout[(size_t)(r0 + i) * NC + col]      = acc0[i];
        Cout[(size_t)(r0 + 16 + i) * NC + col] = acc1[i];
    }
}

// ---------- attention + decoder input pack (one block per batch row) ----------
__global__ __launch_bounds__(256) void attn_step(
        const float* __restrict__ energy,   // [LSRC][NB][HD]
        const u16*   __restrict__ enc_h,    // [LSRC][NB][HD] bf16
        const u16*   __restrict__ h,        // [NB][HD] bf16
        const int*   __restrict__ src,      // [NB][LSRC]
        const int*   __restrict__ tgt,      // [NB][LTGT]
        const float* __restrict__ dec_embed,
        u16* __restrict__ x_dec,            // [NB][ED+HD] bf16
        int t) {
    int n = blockIdx.x;
    int tid = threadIdx.x;
    __shared__ float sc[LSRC];
    __shared__ float part[LSRC][4];

    // scores[l] = energy[l,n,:] . h[n,:]  (4 threads per l, 256 elems each)
    int l = tid >> 2, sub = tid & 3;
    const float* en = energy + ((size_t)l * NB + n) * HD;
    const u16*   hp = h + (size_t)n * HD;
    float s = 0.0f;
    for (int k = sub * 256; k < sub * 256 + 256; ++k) s += en[k] * bf2f(hp[k]);
    part[l][sub] = s;
    __syncthreads();

    if (tid < LSRC) {
        float v = part[tid][0] + part[tid][1] + part[tid][2] + part[tid][3];
        if (src[n * LSRC + tid] == 0) v = -1e9f;   // PAD mask
        sc[tid] = v;
    }
    __syncthreads();
    if (tid == 0) {   // tiny (64-wide) softmax
        float mx = sc[0];
        for (int i = 1; i < LSRC; ++i) mx = fmaxf(mx, sc[i]);
        float sum = 0.0f;
        for (int i = 0; i < LSRC; ++i) { sc[i] = __expf(sc[i] - mx); sum += sc[i]; }
        float inv = 1.0f / sum;
        for (int i = 0; i < LSRC; ++i) sc[i] *= inv;
    }
    __syncthreads();

    // ctx and pack into x_dec[:, ED:]
    for (int hh = tid; hh < HD; hh += 256) {
        float acc = 0.0f;
        for (int l2 = 0; l2 < LSRC; ++l2)
            acc += sc[l2] * bf2f(enc_h[((size_t)l2 * NB + n) * HD + hh]);
        x_dec[(size_t)n * (ED + HD) + ED + hh] = f2bf(acc);
    }
    // embedding into x_dec[:, :ED]
    int tok = tgt[n * LTGT + t];
    for (int e = tid; e < ED; e += 256)
        x_dec[(size_t)n * (ED + HD) + e] = f2bf(dec_embed[(size_t)tok * ED + e]);
}

// ---------- log_softmax over VTGT, one block per batch row ----------
__global__ __launch_bounds__(256) void logsoftmax_k(const float* __restrict__ logits,
                                                    float* __restrict__ out, int t) {
    int n = blockIdx.x, tid = threadIdx.x;
    __shared__ float red[256];
    const float* row = logits + (size_t)n * VTGT;

    float mx = -3.4e38f;
    for (int v = tid; v < VTGT; v += 256) mx = fmaxf(mx, row[v]);
    red[tid] = mx; __syncthreads();
    for (int s = 128; s > 0; s >>= 1) {
        if (tid < s) red[tid] = fmaxf(red[tid], red[tid + s]);
        __syncthreads();
    }
    mx = red[0]; __syncthreads();

    float sum = 0.0f;
    for (int v = tid; v < VTGT; v += 256) sum += __expf(row[v] - mx);
    red[tid] = sum; __syncthreads();
    for (int s = 128; s > 0; s >>= 1) {
        if (tid < s) red[tid] += red[tid + s];
        __syncthreads();
    }
    float lse = mx + logf(red[0]);

    float* o = out + (size_t)n * (LTGT - 1) * VTGT + (size_t)t * VTGT;
    for (int v = tid; v < VTGT; v += 256) o[v] = row[v] - lse;
}

// ====================== host launcher ======================
extern "C" void kernel_launch(void* const* d_in, const int* in_sizes, int n_in,
                              void* d_out, int out_size, void* d_ws, size_t ws_size,
                              hipStream_t stream) {
    (void)in_sizes; (void)n_in; (void)out_size; (void)ws_size;
    const int*   src       = (const int*)d_in[0];
    const int*   tgt       = (const int*)d_in[1];
    const float* enc_embed = (const float*)d_in[2];
    const float* enc_Wi    = (const float*)d_in[3];
    const float* enc_Wh    = (const float*)d_in[4];
    const float* enc_b     = (const float*)d_in[5];
    const float* dec_embed = (const float*)d_in[6];
    const float* dec_Wi    = (const float*)d_in[7];
    const float* dec_Wh    = (const float*)d_in[8];
    const float* dec_b     = (const float*)d_in[9];
    const float* attn_W    = (const float*)d_in[10];
    const float* attn_b    = (const float*)d_in[11];
    const float* h2o_W     = (const float*)d_in[12];
    const float* h2o_b     = (const float*)d_in[13];
    float* out = (float*)d_out;

    char* ws = (char*)d_ws;
    size_t off = 0;
    auto alloc = [&](size_t bytes) -> void* {
        void* p = ws + off;
        off = (off + bytes + 255) & ~(size_t)255;
        return p;
    };
    u16* enc_Wi_p = (u16*)alloc((size_t)ED * G4H * 2);            // 4 MB
    u16* enc_Wh_p = (u16*)alloc((size_t)HD * G4H * 2);            // 8 MB
    u16* dec_Wi_p = (u16*)alloc((size_t)(ED + HD) * G4H * 2);     // 12 MB
    u16* dec_Wh_p = (u16*)alloc((size_t)HD * G4H * 2);            // 8 MB
    u16* attn_W_p = (u16*)alloc((size_t)HD * HD * 2);             // 2 MB
    u16* h2o_p    = (u16*)alloc((size_t)HD * VTGT * 2);           // 64 MB (L2-resident)
    u16* x_src    = (u16*)alloc((size_t)LSRC * NB * ED * 2);      // 2 MB
    u16* enc_h    = (u16*)alloc((size_t)LSRC * NB * HD * 2);      // 4 MB
    float* energy = (float*)alloc((size_t)LSRC * NB * HD * 4);    // 8 MB
    u16* hbuf0    = (u16*)alloc((size_t)NB * HD * 2);
    u16* hbuf1    = (u16*)alloc((size_t)NB * HD * 2);
    float* cbuf   = (float*)alloc((size_t)NB * HD * 4);
    u16* x_dec    = (u16*)alloc((size_t)NB * (ED + HD) * 2);
    float* logits = (float*)alloc((size_t)NB * VTGT * 4);         // 4 MB
    u16* hb[2] = { hbuf0, hbuf1 };

    auto pack = [&](const float* W, u16* dst, int K, int NC) {
        size_t total = (size_t)K * NC;
        pack_w<<<(unsigned)((total + 255) / 256), 256, 0, stream>>>(W, dst, K, NC);
    };
    pack(enc_Wi, enc_Wi_p, ED, G4H);
    pack(enc_Wh, enc_Wh_p, HD, G4H);
    pack(dec_Wi, dec_Wi_p, ED + HD, G4H);
    pack(dec_Wh, dec_Wh_p, HD, G4H);
    pack(attn_W, attn_W_p, HD, HD);
    pack(h2o_W,  h2o_p,    HD, VTGT);

    hipMemsetAsync(hbuf0, 0, (size_t)NB * HD * 2, stream);
    hipMemsetAsync(hbuf1, 0, (size_t)NB * HD * 2, stream);
    hipMemsetAsync(cbuf,  0, (size_t)NB * HD * 4, stream);

    embed_src_k<<<(unsigned)(((size_t)LSRC * NB * ED + 255) / 256), 256, 0, stream>>>(
        src, enc_embed, x_src);

    // ---- encoder: 64 sequential fused LSTM steps ----
    for (int t = 0; t < LSRC; ++t) {
        lstm_step<<<HD / 32, 256, 0, stream>>>(
            x_src + (size_t)t * NB * ED, hb[t & 1],
            enc_Wi_p, enc_Wh_p, enc_b, cbuf,
            hb[(t + 1) & 1], enc_h + (size_t)t * NB * HD, ED);
    }
    int par = 0;   // after 64 steps, current h lives in hb[0]

    // ---- energy = enc_h @ attn_W + attn_b (hoisted out of decoder loop) ----
    {
        int tasks = (LSRC * NB / 32) * (HD / 16);   // 64 * 64 = 4096 wave-tasks
        gemm_bf16<<<tasks / 8, 256, 0, stream>>>(
            enc_h, LSRC * NB, HD, attn_W_p, HD, attn_b, energy);
    }

    // ---- decoder: 47 sequential steps ----
    for (int t = 0; t < LTGT - 1; ++t) {
        attn_step<<<NB, 256, 0, stream>>>(
            energy, enc_h, hb[par], src, tgt, dec_embed, x_dec, t);
        lstm_step<<<HD / 32, 256, 0, stream>>>(
            x_dec, hb[par], dec_Wi_p, dec_Wh_p, dec_b, cbuf,
            hb[par ^ 1], nullptr, ED + HD);
        {
            int tasks = (NB / 32) * (VTGT / 16);    // 1 * 2000 wave-tasks
            gemm_bf16<<<tasks / 8, 256, 0, stream>>>(
                hb[par ^ 1], NB, HD, h2o_p, VTGT, h2o_b, logits);
        }
        logsoftmax_k<<<NB, 256, 0, stream>>>(logits, out, t);
        par ^= 1;
    }
}
